// GCN_16363825397808
// MI455X (gfx1250) — matrix operand order repaired and checked by reading, not compiled, not run on hardware
//
#include <hip/hip_runtime.h>
#include <hip/hip_bf16.h>

typedef __attribute__((ext_vector_type(16))) _Float16 v16h;
typedef __attribute__((ext_vector_type(8)))  _Float16 v8h;
typedef __attribute__((ext_vector_type(4)))  _Float16 v4h;
typedef __attribute__((ext_vector_type(8)))  float    v8f;

#define NN   100000
#define EE   1600000
#define INC  128
#define HH   128
#define CC   64
#define BNEPS 1e-5f

// ---------------------------------------------------------------------------
// Prep kernels
// ---------------------------------------------------------------------------

// W [K x J] f32 row-major  ->  Wt [J x K] f16 (column-contiguous for WMMA B)
__global__ void k_w_to_f16t(const float* __restrict__ W, _Float16* __restrict__ Wt,
                            int K, int J) {
    int idx = blockIdx.x * blockDim.x + threadIdx.x;
    if (idx >= K * J) return;
    int k = idx / J, j = idx % J;
    Wt[j * K + k] = (_Float16)W[idx];
}

// fold eval-mode BN + preceding bias into per-channel scale/shift
__global__ void k_fold_bn(const float* __restrict__ g, const float* __restrict__ beta,
                          const float* __restrict__ rm, const float* __restrict__ rv,
                          const float* __restrict__ bias,
                          float* __restrict__ s, float* __restrict__ t, int n) {
    int j = blockIdx.x * blockDim.x + threadIdx.x;
    if (j >= n) return;
    float sj = g[j] * rsqrtf(rv[j] + BNEPS);
    s[j] = sj;
    t[j] = (bias[j] - rm[j]) * sj + beta[j];
}

__global__ void k_fill(float* __restrict__ p, float v, int n) {
    int i = blockIdx.x * blockDim.x + threadIdx.x;
    if (i < n) p[i] = v;
}

__global__ void k_deg(const int* __restrict__ cols, float* __restrict__ deg, int e) {
    int i = blockIdx.x * blockDim.x + threadIdx.x;
    if (i < e) atomicAdd(&deg[cols[i]], 1.0f);
}

__global__ void k_rsqrt(const float* __restrict__ deg, float* __restrict__ dinv, int n) {
    int i = blockIdx.x * blockDim.x + threadIdx.x;
    if (i < n) dinv[i] = rsqrtf(deg[i]);
}

// ---------------------------------------------------------------------------
// WMMA GEMM: Y[N x KOUT] = X[N x 128] @ W[128 x KOUT], optional BN+ReLU epilogue
// 256 threads = 8 waves; each wave -> one 16x16 output tile; K=128 as 4 x
// v_wmma_f32_16x16x32_f16. Rows/block: 16 (KOUT=128) or 32 (KOUT=64).
// ---------------------------------------------------------------------------
template<int KOUT, bool EPI>
__global__ __launch_bounds__(256) void k_gemm(const float* __restrict__ X,
                                              const _Float16* __restrict__ Wt, // [KOUT][128]
                                              const float* __restrict__ s,
                                              const float* __restrict__ t,
                                              float* __restrict__ Y) {
    constexpr int K    = 128;
    constexpr int RG   = (8 * 16) / KOUT;   // row groups per block
    constexpr int ROWS = 16 * RG;
    constexpr int LDSS = K + 8;             // padded stride (halves) -> conflict-free b128
    __shared__ alignas(16) _Float16 As[ROWS][LDSS];

    const int tid  = threadIdx.x;
    const int row0 = blockIdx.x * ROWS;

    // stage X tile as f16 into LDS (tile is contiguous ROWS*K floats)
    constexpr int F4_PER_T = (ROWS * K / 4) / 256;
    const float4* Xv = (const float4*)(X + (size_t)row0 * K);
    #pragma unroll
    for (int i = 0; i < F4_PER_T; ++i) {
        int e  = tid + i * 256;
        int r  = e / (K / 4);
        int kc = (e % (K / 4)) * 4;
        float4 v = Xv[e];
        v4h hv;
        hv[0] = (_Float16)v.x; hv[1] = (_Float16)v.y;
        hv[2] = (_Float16)v.z; hv[3] = (_Float16)v.w;
        *(v4h*)&As[r][kc] = hv;
    }
    __syncthreads();

    const int wave = tid >> 5;
    const int lane = tid & 31;
    const int h    = lane >> 4;        // lane half selects K sub-block
    const int ln   = lane & 15;        // A row / B column / C column
    const int rg   = wave / (KOUT / 16);
    const int c0   = (wave % (KOUT / 16)) * 16;

    const _Float16* colp = Wt + (size_t)(c0 + ln) * K;   // B column (contiguous K)
    const _Float16* arow = &As[rg * 16 + ln][0];

    v8f acc = {};
    #pragma unroll
    for (int kk = 0; kk < K; kk += 32) {
        // A 16x32 f16 layout: elem e -> K = kk + h*8 + e (e<8), kk + 16 + h*8 + (e-8)
        v8h alo = *(const v8h*)(arow + kk + h * 8);
        v8h ahi = *(const v8h*)(arow + kk + 16 + h * 8);
        v16h a;
        #pragma unroll
        for (int i = 0; i < 8; ++i) { a[i] = alo[i]; a[i + 8] = ahi[i]; }
        // B 32x16 f16 layout: elem e -> K = kk + h*8 + e (e<8), kk + 16 + h*8 + (e-8)
        v8h blo = *(const v8h*)(colp + kk + h * 8);
        v8h bhi = *(const v8h*)(colp + kk + 16 + h * 8);
        v16h b;
        #pragma unroll
        for (int i = 0; i < 8; ++i) { b[i] = blo[i]; b[i + 8] = bhi[i]; }

        acc = __builtin_amdgcn_wmma_f32_16x16x32_f16(false, a, false, b,
                                                     (short)0, acc, false, false);
    }

    // C/D layout: VGPR r -> row (r + 8*h), col ln
    #pragma unroll
    for (int r = 0; r < 8; ++r) {
        int row = row0 + rg * 16 + r + 8 * h;
        int col = c0 + ln;
        float v = acc[r];
        if constexpr (EPI) {
            v = v * s[col] + t[col];
            v = v > 0.f ? v : 0.f;
        }
        Y[(size_t)row * KOUT + col] = v;
    }
}

// ---------------------------------------------------------------------------
// Edge scatter-add: out[c] += msg[r] * dinv[r]*dinv[c]; self loops appended.
// One thread = 4 consecutive features of one message; L2-resident atomics.
// ---------------------------------------------------------------------------
template<int FD>
__global__ __launch_bounds__(256) void k_scatter(const float* __restrict__ msg,
                                                 const int* __restrict__ rows,
                                                 const int* __restrict__ cols,
                                                 const float* __restrict__ dinv,
                                                 float* __restrict__ out,
                                                 int n_edges, int n_total) {
    constexpr int LPE = FD / 4;
    long long idx = (long long)blockIdx.x * blockDim.x + threadIdx.x;
    int e = (int)(idx / LPE);
    if (e >= n_total) return;
    int j = (int)(idx % LPE) * 4;
    int r, c;
    if (e < n_edges) { r = rows[e]; c = cols[e]; }
    else             { r = c = e - n_edges; }
    float w = dinv[r] * dinv[c];
    float4 v = *(const float4*)(msg + (size_t)r * FD + j);
    float* op = out + (size_t)c * FD + j;
    atomicAdd(op + 0, v.x * w);
    atomicAdd(op + 1, v.y * w);
    atomicAdd(op + 2, v.z * w);
    atomicAdd(op + 3, v.w * w);
}

// in-place per-channel scale/shift + ReLU (post-scatter BN)
__global__ void k_bnrelu(float* __restrict__ x, const float* __restrict__ s,
                         const float* __restrict__ t, int total) {
    int i = blockIdx.x * blockDim.x + threadIdx.x;
    if (i >= total) return;
    int j = i & (HH - 1);
    float v = x[i] * s[j] + t[j];
    x[i] = v > 0.f ? v : 0.f;
}

// row-wise log_softmax over 64 classes (+ final conv bias), one wave32 per row
__global__ __launch_bounds__(256) void k_logsoftmax(const float* __restrict__ in,
                                                    const float* __restrict__ bias,
                                                    float* __restrict__ out, int nrows) {
    int row  = blockIdx.x * 8 + (threadIdx.x >> 5);
    int lane = threadIdx.x & 31;
    if (row >= nrows) return;
    const float* p = in + (size_t)row * CC;
    float v0 = p[lane]      + bias[lane];
    float v1 = p[lane + 32] + bias[lane + 32];
    float m = fmaxf(v0, v1);
    #pragma unroll
    for (int off = 16; off > 0; off >>= 1) m = fmaxf(m, __shfl_xor(m, off, 32));
    float ss = expf(v0 - m) + expf(v1 - m);
    #pragma unroll
    for (int off = 16; off > 0; off >>= 1) ss += __shfl_xor(ss, off, 32);
    float l = m + logf(ss);
    float* o = out + (size_t)row * CC;
    o[lane]      = v0 - l;
    o[lane + 32] = v1 - l;
}

// ---------------------------------------------------------------------------
extern "C" void kernel_launch(void* const* d_in, const int* in_sizes, int n_in,
                              void* d_out, int out_size, void* d_ws, size_t ws_size,
                              hipStream_t stream) {
    (void)in_sizes; (void)n_in; (void)out_size; (void)ws_size;

    const float* x      = (const float*)d_in[0];
    const int*   ei     = (const int*)  d_in[1];
    const float* linW1  = (const float*)d_in[2];
    const float* linb1  = (const float*)d_in[3];
    const float* linW2  = (const float*)d_in[4];
    const float* linb2  = (const float*)d_in[5];
    const float* bnG[4], *bnB[4], *bnM[4], *bnV[4];
    for (int i = 0; i < 4; ++i) {
        bnG[i] = (const float*)d_in[6 + 4 * i + 0];
        bnB[i] = (const float*)d_in[6 + 4 * i + 1];
        bnM[i] = (const float*)d_in[6 + 4 * i + 2];
        bnV[i] = (const float*)d_in[6 + 4 * i + 3];
    }
    const float* convW1 = (const float*)d_in[22];
    const float* convb1 = (const float*)d_in[23];
    const float* convW2 = (const float*)d_in[24];
    const float* convb2 = (const float*)d_in[25];
    const float* convW3 = (const float*)d_in[26];
    const float* convb3 = (const float*)d_in[27];
    const int* rows = ei;
    const int* cols = ei + EE;

    // workspace carve-up (256B aligned)
    char* ws = (char*)d_ws;
    size_t off = 0;
    auto alloc = [&](size_t bytes) -> void* {
        void* p = ws + off;
        off += (bytes + 255) & ~(size_t)255;
        return p;
    };
    float*    bufA = (float*)alloc((size_t)NN * HH * 4);
    float*    bufB = (float*)alloc((size_t)NN * HH * 4);
    float*    bufC = (float*)alloc((size_t)NN * CC * 4);
    float*    dinv = (float*)alloc((size_t)NN * 4);
    _Float16* WtL1 = (_Float16*)alloc(128 * 128 * 2);
    _Float16* WtL2 = (_Float16*)alloc(128 * 128 * 2);
    _Float16* WtC1 = (_Float16*)alloc(128 * 128 * 2);
    _Float16* WtC2 = (_Float16*)alloc(128 * 128 * 2);
    _Float16* WtC3 = (_Float16*)alloc(128 * 64 * 2);
    float* sv[4]; float* tv[4];
    for (int i = 0; i < 4; ++i) { sv[i] = (float*)alloc(128 * 4); tv[i] = (float*)alloc(128 * 4); }

    // ---- prep: f16 weight transposes, folded BN affines, symmetric norm ----
    k_w_to_f16t<<<(128 * 128 + 255) / 256, 256, 0, stream>>>(linW1,  WtL1, 128, 128);
    k_w_to_f16t<<<(128 * 128 + 255) / 256, 256, 0, stream>>>(linW2,  WtL2, 128, 128);
    k_w_to_f16t<<<(128 * 128 + 255) / 256, 256, 0, stream>>>(convW1, WtC1, 128, 128);
    k_w_to_f16t<<<(128 * 128 + 255) / 256, 256, 0, stream>>>(convW2, WtC2, 128, 128);
    k_w_to_f16t<<<(128 * 64  + 255) / 256, 256, 0, stream>>>(convW3, WtC3, 128, 64);

    k_fold_bn<<<1, 128, 0, stream>>>(bnG[0], bnB[0], bnM[0], bnV[0], linb1,  sv[0], tv[0], 128);
    k_fold_bn<<<1, 128, 0, stream>>>(bnG[1], bnB[1], bnM[1], bnV[1], linb2,  sv[1], tv[1], 128);
    k_fold_bn<<<1, 128, 0, stream>>>(bnG[2], bnB[2], bnM[2], bnV[2], convb1, sv[2], tv[2], 128);
    k_fold_bn<<<1, 128, 0, stream>>>(bnG[3], bnB[3], bnM[3], bnV[3], convb2, sv[3], tv[3], 128);

    float* deg = bufC;  // temp reuse (bufC re-zeroed before conv3 scatter)
    k_fill <<<(NN + 255) / 256, 256, 0, stream>>>(deg, 1.0f, NN);   // self loop
    k_deg  <<<(EE + 255) / 256, 256, 0, stream>>>(cols, deg, EE);
    k_rsqrt<<<(NN + 255) / 256, 256, 0, stream>>>(deg, dinv, NN);

    const int G128 = NN / 16;   // 6250
    const int G64  = NN / 32;   // 3125

    // ---- input MLP: Linear -> BN -> ReLU  (x2) ----
    k_gemm<128, true ><<<G128, 256, 0, stream>>>(x,    WtL1, sv[0], tv[0], bufA);
    k_gemm<128, true ><<<G128, 256, 0, stream>>>(bufA, WtL2, sv[1], tv[1], bufB);

    // ---- conv1: dense transform, scatter-add, BN+ReLU ----
    k_gemm<128, false><<<G128, 256, 0, stream>>>(bufB, WtC1, nullptr, nullptr, bufA);
    hipMemsetAsync(bufB, 0, (size_t)NN * HH * 4, stream);
    {
        long long thr = (long long)(EE + NN) * (HH / 4);
        k_scatter<128><<<(int)((thr + 255) / 256), 256, 0, stream>>>(
            bufA, rows, cols, dinv, bufB, EE, EE + NN);
    }
    k_bnrelu<<<(NN * HH + 255) / 256, 256, 0, stream>>>(bufB, sv[2], tv[2], NN * HH);

    // ---- conv2 ----
    k_gemm<128, false><<<G128, 256, 0, stream>>>(bufB, WtC2, nullptr, nullptr, bufA);
    hipMemsetAsync(bufB, 0, (size_t)NN * HH * 4, stream);
    {
        long long thr = (long long)(EE + NN) * (HH / 4);
        k_scatter<128><<<(int)((thr + 255) / 256), 256, 0, stream>>>(
            bufA, rows, cols, dinv, bufB, EE, EE + NN);
    }
    k_bnrelu<<<(NN * HH + 255) / 256, 256, 0, stream>>>(bufB, sv[3], tv[3], NN * HH);

    // ---- conv3 (out_channels=64) + log_softmax ----
    k_gemm<64, false><<<G64, 256, 0, stream>>>(bufB, WtC3, nullptr, nullptr, bufA);
    hipMemsetAsync(bufC, 0, (size_t)NN * CC * 4, stream);
    {
        long long thr = (long long)(EE + NN) * (CC / 4);
        k_scatter<64><<<(int)((thr + 255) / 256), 256, 0, stream>>>(
            bufA, rows, cols, dinv, bufC, EE, EE + NN);
    }
    k_logsoftmax<<<NN / 8, 256, 0, stream>>>(bufC, convb3, (float*)d_out, NN);
}
